// unit_vit_27986006901254
// MI455X (gfx1250) — matrix operand order repaired and checked by reading, not compiled, not run on hardware
//
#include <hip/hip_runtime.h>
#include <hip/hip_bf16.h>

typedef _Float16 half_t;
typedef __attribute__((ext_vector_type(16))) _Float16 v16h;
typedef __attribute__((ext_vector_type(8)))  _Float16 v8h;
typedef __attribute__((ext_vector_type(8)))  float    v8f;
typedef unsigned int u32x4 __attribute__((ext_vector_type(4)));
typedef int          i32x4 __attribute__((ext_vector_type(4)));
typedef int          i32x8 __attribute__((ext_vector_type(8)));

// ---------------- problem constants ----------------
constexpr int cN = 16, cC = 256, cT = 256, cV = 25, cH = 8;
constexpr int cDIM = 256, cHD = 32, cINTER = 64, cG = 5;
constexpr int cNT  = cN * cT;          // 4096
constexpr int cNTV = cNT * cV;         // 102400
constexpr int cNTG = cNT * cG;         // 20480
constexpr int cK   = 256;              // reduction dim of all big GEMMs
constexpr int cTS  = 1024;             // padded per-(nt,h) tile stride (32 rows x 32)
constexpr float cSCALE = 0.17677669529663687f;   // 32^-0.5
constexpr float cLAMBDA_INIT = 0.2f;

#define WMMA_F16(a, b, c) \
  __builtin_amdgcn_wmma_f32_16x16x32_f16(false, (a), false, (b), (short)0, (c), false, false)

__device__ __forceinline__ v16h cat8(v8h a, v8h b) {
  return __builtin_shufflevector(a, b, 0, 1, 2, 3, 4, 5, 6, 7, 8, 9, 10, 11, 12, 13, 14, 15);
}

// ---------------- WMMA fragment helpers (all unconditional, vectorized) ----------------
// A fragment, row-major A[row][k] (lda halves), 16x32 tile at (row0, k0). Rows must be valid.
__device__ __forceinline__ v16h load_a_frag(const half_t* __restrict__ A, int lda,
                                            int row0, int k0, int lane) {
  int hi = (lane & 16) ? 1 : 0;
  const half_t* ap = A + (size_t)(row0 + (lane & 15)) * lda;
  int ka = k0 + (hi ? 16 : 0);
  int kb = k0 + (hi ? 24 : 8);
  return cat8(*(const v8h*)(ap + ka), *(const v8h*)(ap + kb));
}

// A fragment of the transpose of a [32][32] tile: A^T[row=d][k=m] = tile[m*32+d].
__device__ __forceinline__ v16h load_a_frag_T(const half_t* __restrict__ tile,
                                              int row0, int lane) {
  int hi = (lane & 16) ? 1 : 0;
  int d  = row0 + (lane & 15);
  int m0 = hi ? 16 : 0, m1 = hi ? 24 : 8;
  v16h f;
#pragma unroll
  for (int i = 0; i < 8; ++i) f[i]     = tile[(m0 + i) * 32 + d];
#pragma unroll
  for (int i = 0; i < 8; ++i) f[8 + i] = tile[(m1 + i) * 32 + d];
  return f;
}

// B fragment from column-major B[col][K]: element (k, col) = B[col*K + k].
__device__ __forceinline__ v16h load_b_frag_cm(const half_t* __restrict__ B,
                                               int k0, int col0, int lane) {
  const half_t* bp = B + (size_t)(col0 + (lane & 15)) * cK + k0 + ((lane & 16) ? 16 : 0);
  return cat8(*(const v8h*)bp, *(const v8h*)(bp + 8));
}

// B fragment of the transpose of a [32][32] tile: B[k=d][n] = tile[n*32+d]. K=32 fixed.
__device__ __forceinline__ v16h load_b_frag_tile(const half_t* __restrict__ tile, int col0,
                                                 int lane) {
  const half_t* tp = tile + (size_t)(col0 + (lane & 15)) * 32 + ((lane & 16) ? 16 : 0);
  return cat8(*(const v8h*)tp, *(const v8h*)(tp + 8));
}

// Re-pack two C/D accumulators (M-tiles 0/1 of a 32xN^T result) into a B fragment.
__device__ __forceinline__ v16h frag_from_accT(v8f a0, v8f a1, int lane, float scale) {
  v16h f;
#pragma unroll
  for (int i = 0; i < 8; ++i) {
    float o0 = __shfl_xor(a0[i] * scale, 16);
    float o1 = __shfl_xor(a1[i] * scale, 16);
    float s0 = a0[i] * scale, s1 = a1[i] * scale;
    if (lane < 16) { f[i] = (half_t)s0; f[8 + i] = (half_t)o0; }
    else           { f[i] = (half_t)o1; f[8 + i] = (half_t)s1; }
  }
  return f;
}

// ---------------- generic WMMA GEMM: Y[o,j] = A[o,:] * B[:,j], K = 256 ----------------
// Block = 128-row A panel (staged to LDS via TDM) x 64 j columns; 8 waves,
// each wave owns one 16-row tile x four independent 16-col accumulators
// (the four WMMAs per K-step fill each other's RAW-hazard slots).
// mode 0: f32 store out[o*ncols+j]
// mode 1: q/k scatter f16 (M=512): o=(s,h,d) -> tile[((nt*2+s)*8+h)*1024 + v*32+d]
// mode 2: v/ek scatter f16 (M=256): o=(h,d)  -> tile[(nt*8+h)*1024 + v*32+d]
// mode 3: f32 accumulate (read-modify-write) + bias[o]
__global__ __launch_bounds__(256) void gemm_wmma_kernel(
    const half_t* __restrict__ A, const half_t* __restrict__ Bcm,
    int M, int ncols, int mode,
    float* __restrict__ outf, half_t* __restrict__ outh,
    const float* __restrict__ bias) {
  __shared__ __align__(16) half_t As[128 * cK];   // 64 KB A panel

  int mpanels = M >> 7;
  int oPanel = blockIdx.x % mpanels;
  int jTile  = blockIdx.x / mpanels;
  int wv = threadIdx.x >> 5, lane = threadIdx.x & 31;
  const half_t* Apanel = A + (size_t)oPanel * 128 * cK;

#if defined(__has_builtin) && __has_builtin(__builtin_amdgcn_tensor_load_to_lds)
  if (threadIdx.x < 32) {
    // Tensor DMA descriptor (D#): 2-D tile, data_size=2B, 256 x 128 elements.
    size_t gaddr = (size_t)Apanel;
    u32x4 g0;
    g0[0] = 1u;                                             // count=1, user mode
    g0[1] = 0u;                                             // lds_addr = 0 (As at base)
    g0[2] = (unsigned)(gaddr & 0xffffffffu);                // global_addr[31:0]
    g0[3] = (unsigned)((gaddr >> 32) & 0x01ffffffu) | (2u << 30);  // addr[56:32] | type=2
    unsigned long long q0 = 0x10000ull |                    // data_size=1 (2 bytes)
                            ((unsigned long long)(cK & 0xffff) << 48);   // tensor_dim0 lo
    unsigned long long q1 = (unsigned long long)(cK >> 16) |             // tensor_dim0 hi
                            ((unsigned long long)128u << 16) |           // tensor_dim1
                            ((unsigned long long)(cK & 0xffff) << 48);   // tile_dim0
    unsigned long long q2 = (unsigned long long)128u |                   // tile_dim1
                            ((unsigned long long)(unsigned)cK << 32);    // dim0_stride lo
    unsigned long long q3 = 0ull;                                        // dim0_stride hi
    i32x8 g1;
    g1[0] = (int)q0; g1[1] = (int)(q0 >> 32);
    g1[2] = (int)q1; g1[3] = (int)(q1 >> 32);
    g1[4] = (int)q2; g1[5] = (int)(q2 >> 32);
    g1[6] = (int)q3; g1[7] = (int)(q3 >> 32);
    i32x4 gz = {0, 0, 0, 0};
    i32x8 gz8 = {0, 0, 0, 0, 0, 0, 0, 0};
    __builtin_amdgcn_tensor_load_to_lds(g0, g1, gz, gz, gz8, 0);
    __builtin_amdgcn_s_wait_tensorcnt(0);
  }
#else
  for (int i = threadIdx.x; i < 128 * cK / 8; i += 256)
    ((v8h*)As)[i] = ((const v8h*)Apanel)[i];
#endif
  __syncthreads();

  int o0l = wv << 4;                 // local row base within panel
  int o0  = oPanel * 128 + o0l;
  int j0  = jTile << 6;              // 64 columns per block
  int rb  = (lane & 16) ? 8 : 0;

#pragma unroll
  for (int t2 = 0; t2 < 4; ++t2)
    __builtin_prefetch(Bcm + (size_t)(j0 + t2 * 16 + (lane & 15)) * cK, 0, 1);

  v8f accs[4];
  accs[0] = {}; accs[1] = {}; accs[2] = {}; accs[3] = {};
  if (mode == 3) {
#pragma unroll
    for (int t2 = 0; t2 < 4; ++t2) {
      int col = j0 + t2 * 16 + (lane & 15);
#pragma unroll
      for (int r = 0; r < 8; ++r) accs[t2][r] = outf[(size_t)(o0 + rb + r) * ncols + col];
    }
  }

#pragma unroll
  for (int k0 = 0; k0 < cK; k0 += 32) {
    v16h a = load_a_frag(As, cK, o0l, k0, lane);      // LDS (ds_load_b128)
    v16h b0 = load_b_frag_cm(Bcm, k0, j0,      lane); // global (b128 x2 each)
    v16h b1 = load_b_frag_cm(Bcm, k0, j0 + 16, lane);
    v16h b2 = load_b_frag_cm(Bcm, k0, j0 + 32, lane);
    v16h b3 = load_b_frag_cm(Bcm, k0, j0 + 48, lane);
    accs[0] = WMMA_F16(a, b0, accs[0]);
    accs[1] = WMMA_F16(a, b1, accs[1]);
    accs[2] = WMMA_F16(a, b2, accs[2]);
    accs[3] = WMMA_F16(a, b3, accs[3]);
  }

#pragma unroll
  for (int t2 = 0; t2 < 4; ++t2) {
    int col = j0 + t2 * 16 + (lane & 15);
    if (mode == 0) {
#pragma unroll
      for (int r = 0; r < 8; ++r) outf[(size_t)(o0 + rb + r) * ncols + col] = accs[t2][r];
    } else if (mode == 1) {
      int nt = col / cV, vv = col % cV;
#pragma unroll
      for (int r = 0; r < 8; ++r) {
        int o = o0 + rb + r;
        int s = o >> 8, h = (o >> 5) & 7, d = o & 31;
        outh[(((size_t)nt * 2 + s) * 8 + h) * cTS + vv * 32 + d] = (half_t)accs[t2][r];
      }
    } else if (mode == 2) {
      int nt = col / cV, vv = col % cV;
#pragma unroll
      for (int r = 0; r < 8; ++r) {
        int o = o0 + rb + r;
        int h = o >> 5, d = o & 31;
        outh[((size_t)nt * 8 + h) * cTS + vv * 32 + d] = (half_t)accs[t2][r];
      }
    } else {
#pragma unroll
      for (int r = 0; r < 8; ++r) {
        int o = o0 + rb + r;
        outf[(size_t)o * ncols + col] = accs[t2][r] + bias[o];
      }
    }
  }
}

// ---------------- LayerNorm over C, emit f16 column-major [NTV][C] ----------------
__global__ void ln_kernel(const float* __restrict__ x, const float* __restrict__ lw,
                          const float* __restrict__ lb, half_t* __restrict__ xn) {
  int wave = (blockIdx.x * blockDim.x + threadIdx.x) >> 5;
  int lane = threadIdx.x & 31;
  if (wave >= cNTV) return;
  int j = wave;
  int n = j / (cT * cV), rem = j % (cT * cV);
  const float* xp = x + (size_t)n * cC * cT * cV + rem;
  float vals[8], s = 0.f, ss = 0.f;
#pragma unroll
  for (int i = 0; i < 8; ++i) {
    float t = xp[(size_t)(lane + i * 32) * cT * cV];
    vals[i] = t; s += t; ss += t * t;
  }
#pragma unroll
  for (int off = 16; off; off >>= 1) { s += __shfl_xor(s, off); ss += __shfl_xor(ss, off); }
  float mu = s * (1.f / cC);
  float var = ss * (1.f / cC) - mu * mu;
  float rstd = rsqrtf(var + 1e-5f);
#pragma unroll
  for (int i = 0; i < 8; ++i) {
    int c = lane + i * 32;
    xn[(size_t)j * cC + c] = (half_t)(((vals[i] - mu) * rstd) * lw[c] + lb[c]);
  }
}

// ---------------- hyperedge pooling: z16 column-major [NTG][C] ----------------
__global__ void poolz_kernel(const float* __restrict__ x, const int* __restrict__ jl,
                             half_t* __restrict__ z16) {
  int idx = blockIdx.x * blockDim.x + threadIdx.x;
  if (idx >= cC * cNTG) return;
  int c = idx % cC, col = idx / cC;
  int g = col % cG, nt = col / cG;
  int n = nt / cT, t = nt % cT;
  const float* xp = x + (((size_t)n * cC + c) * cT + t) * cV;
  float cnt = 1e-6f, s = 0.f;
#pragma unroll
  for (int v = 0; v < cV; ++v) {
    if (jl[v] == g) { cnt += 1.f; s += xp[v]; }
  }
  z16[(size_t)col * cC + c] = (half_t)(s / cnt);
}

// ---------------- zero the 7 pad rows (25..31) of each 32x32 tile ----------------
__global__ void zeropad_kernel(half_t* __restrict__ p, int ntiles) {
  int idx = blockIdx.x * blockDim.x + threadIdx.x;
  if (idx >= ntiles * 224) return;
  int tile = idx / 224, r = idx % 224;
  p[(size_t)tile * cTS + 800 + r] = (half_t)0;
}

// ---------------- gather e back per joint + d-term ----------------
__global__ void egather_kernel(const float* __restrict__ eo, const int* __restrict__ jl,
                               const float* __restrict__ w1,
                               half_t* __restrict__ ek, float* __restrict__ dterm) {
  int idx = blockIdx.x * blockDim.x + threadIdx.x;
  if (idx >= cNT * cH * cV) return;
  int vv = idx % cV, rem = idx / cV;
  int h = rem % cH, nt = rem / cH;
  int colg = nt * cG + jl[vv];
  half_t* dst = ek + ((size_t)nt * cH + h) * cTS + vv * 32;
  float ds = 0.f;
#pragma unroll
  for (int d = 0; d < cHD; ++d) {
    float val = eo[(size_t)(h * cHD + d) * cNTG + colg];
    dst[d] = (half_t)val;
    ds += w1[h * cHD + d] * val;
  }
  dterm[((size_t)nt * cH + h) * cV + vv] = ds;
}

// ---------------- lambda scalar ----------------
__global__ void lam_kernel(const float* lq1, const float* lk1,
                           const float* lq2, const float* lk2, float* lam) {
  float s1 = 0.f, s2 = 0.f;
#pragma unroll
  for (int i = 0; i < 16; ++i) { s1 += lq1[i] * lk1[i]; s2 += lq2[i] * lk2[i]; }
  lam[0] = __expf(s1) - __expf(s2) + cLAMBDA_INIT;
}

// ---------------- differential attention + grouped proj ----------------
__global__ __launch_bounds__(256) void attn_kernel(
    const half_t* __restrict__ q, const half_t* __restrict__ k,
    const half_t* __restrict__ v, const half_t* __restrict__ ek,
    const float* __restrict__ dterm, const float* __restrict__ rpe,
    const int* __restrict__ hops, const float* __restrict__ outer,
    const float* __restrict__ alpha_p, const float* __restrict__ lam_p,
    const half_t* __restrict__ wp, const float* __restrict__ bproj,
    float* __restrict__ xtr) {
  int nt   = blockIdx.x;
  int h    = threadIdx.x >> 5;
  int lane = threadIdx.x & 31;
  float alpha = alpha_p[0];
  float lam   = lam_p[0];

  const half_t* vt  = v  + ((size_t)nt * cH + h) * cTS;
  const half_t* ekt = ek + ((size_t)nt * cH + h) * cTS;
  const float*  dt  = dterm + ((size_t)nt * cH + h) * cV;
  const float*  oub = outer + (size_t)h * cV * cV;

  v8f oav[2][2];
  oav[0][0] = {}; oav[0][1] = {}; oav[1][0] = {}; oav[1][1] = {};

  int mlo = (lane & 16) ? 8 : 0;

  for (int s = 0; s < 2; ++s) {
    const half_t* qt = q + (((size_t)nt * 2 + s) * cH + h) * cTS;
    const half_t* kt = k + (((size_t)nt * 2 + s) * cH + h) * cTS;

    // ---- logits^T[m,n] = K*Q^T + EK*Q^T  (a- and c-terms; pads are zero) ----
    v16h bq0 = load_b_frag_tile(qt, 0,  lane);
    v16h bq1 = load_b_frag_tile(qt, 16, lane);
    v8f accv[2][2];
#pragma unroll
    for (int mt = 0; mt < 2; ++mt) {
      v16h ak = load_a_frag(kt,  32, mt * 16, 0, lane);
      v16h ae = load_a_frag(ekt, 32, mt * 16, 0, lane);
      v8f t0 = {}; t0 = WMMA_F16(ak, bq0, t0); t0 = WMMA_F16(ae, bq0, t0);
      v8f t1 = {}; t1 = WMMA_F16(ak, bq1, t1); t1 = WMMA_F16(ae, bq1, t1);
      accv[mt][0] = t0; accv[mt][1] = t1;
    }

    // ---- b-term (rel-pos) + d-term + scale + softmax + P^T ----
    float fscale = (s == 0) ? 1.f : -lam;
#pragma unroll
    for (int ncol = 0; ncol < 2; ++ncol) {
      int nidx = ncol * 16 + (lane & 15);
      bool nvalid = nidx < cV;
      int ni = nvalid ? nidx : 0;
      float qf[cHD];
#pragma unroll
      for (int kk = 0; kk < cHD; ++kk) qf[kk] = (float)qt[nidx * 32 + kk]; // pad rows = 0

#pragma unroll
      for (int mt = 0; mt < 2; ++mt) {
        int mb = mt * 16 + mlo;
#pragma unroll
        for (int r = 0; r < 8; ++r) {
          int m = mb + r;
          bool mvalid = m < cV;
          int mi = mvalid ? m : 0;
          const float* kr = rpe + (size_t)hops[ni * cV + mi] * cDIM + h * cHD;
          float su = 0.f;
#pragma unroll
          for (int kk = 0; kk < cHD; ++kk) su += qf[kk] * kr[kk];
          float add = (nvalid && mvalid) ? (su + dt[mi]) : 0.f;
          accv[mt][ncol][r] = (accv[mt][ncol][r] + add) * cSCALE;
        }
      }

      // softmax over m for column n (lane pair via shfl_xor 16)
      float mx = -1e30f;
#pragma unroll
      for (int mt = 0; mt < 2; ++mt)
#pragma unroll
        for (int r = 0; r < 8; ++r) {
          int m = mt * 16 + mlo + r;
          if (m < cV) mx = fmaxf(mx, accv[mt][ncol][r]);
        }
      mx = fmaxf(mx, __shfl_xor(mx, 16));
      float su = 0.f;
#pragma unroll
      for (int mt = 0; mt < 2; ++mt)
#pragma unroll
        for (int r = 0; r < 8; ++r) {
          int m = mt * 16 + mlo + r;
          float e_ = (m < cV) ? __expf(accv[mt][ncol][r] - mx) : 0.f;
          accv[mt][ncol][r] = e_; su += e_;
        }
      su += __shfl_xor(su, 16);
      float inv = (nvalid && su > 0.f) ? 1.f / su : 0.f;

      // P^T = fscale * (alpha * attn^T + outer^T); zero the pads
#pragma unroll
      for (int mt = 0; mt < 2; ++mt)
#pragma unroll
        for (int r = 0; r < 8; ++r) {
          int m = mt * 16 + mlo + r;
          int mi = (m < cV) ? m : 0;
          float ov = oub[ni * cV + mi];
          float Pv = (nvalid && m < cV) ? fscale * (alpha * accv[mt][ncol][r] * inv + ov) : 0.f;
          accv[mt][ncol][r] = Pv;
        }
    }

    // ---- out^T += V^T * P^T ----
#pragma unroll
    for (int ncol = 0; ncol < 2; ++ncol) {
      v16h bp = frag_from_accT(accv[0][ncol], accv[1][ncol], lane, 1.f);
#pragma unroll
      for (int dt2 = 0; dt2 < 2; ++dt2) {
        v16h av = load_a_frag_T(vt, dt2 * 16, lane);
        oav[dt2][ncol] = WMMA_F16(av, bp, oav[dt2][ncol]);
      }
    }
  }

  // ---- grouped projection: y^T = Wp[h] * out^T, add bias, store x_tr ----
  const half_t* wph = wp + (size_t)h * cHD * cHD;
#pragma unroll
  for (int ncol = 0; ncol < 2; ++ncol) {
    v16h bx = frag_from_accT(oav[0][ncol], oav[1][ncol], lane, 1.f);
#pragma unroll
    for (int ot = 0; ot < 2; ++ot) {
      v16h aw = load_a_frag(wph, 32, ot * 16, 0, lane);
      v8f y = {};
      y = WMMA_F16(aw, bx, y);
      int vv = ncol * 16 + (lane & 15);
      if (vv < cV) {
        size_t jj = (size_t)nt * cV + vv;
        int rbl = ot * 16 + ((lane & 16) ? 8 : 0);
#pragma unroll
        for (int r = 0; r < 8; ++r) {
          int cch = h * cHD + rbl + r;
          xtr[(size_t)cch * cNTV + jj] = y[r] + bproj[cch];
        }
      }
    }
  }
}

// ---------------- GCN helpers ----------------
__global__ void xmean_kernel(const float* __restrict__ x, float* __restrict__ xm) {
  int idx = blockIdx.x * blockDim.x + threadIdx.x;
  if (idx >= cN * cC * cV) return;
  int vv = idx % cV, rem = idx / cV;
  int c = rem % cC, n = rem / cC;
  const float* xp = x + (((size_t)n * cC + c) * cT) * cV + vv;
  float s = 0.f;
  for (int t = 0; t < cT; ++t) s += xp[(size_t)t * cV];
  xm[idx] = s * (1.f / cT);
}

__global__ void thph_kernel(const float* __restrict__ xm,
                            const float* __restrict__ Wth, const float* __restrict__ bth,
                            const float* __restrict__ Wph, const float* __restrict__ bph,
                            float* __restrict__ th, float* __restrict__ ph) {
  int idx = blockIdx.x * blockDim.x + threadIdx.x;
  if (idx >= cN * cINTER * cV) return;
  int vv = idx % cV, rem = idx / cV;
  int i = rem % cINTER, n = rem / cINTER;
  const float* xv = xm + (size_t)n * cC * cV + vv;
  float s1 = bth[i], s2 = bph[i];
  for (int c = 0; c < cC; ++c) {
    float xx = xv[(size_t)c * cV];
    s1 += Wth[i * cC + c] * xx;
    s2 += Wph[i * cC + c] * xx;
  }
  th[idx] = s1; ph[idx] = s2;
}

__global__ void scoresE_kernel(const float* __restrict__ th, const float* __restrict__ ph,
                               float* __restrict__ E) {
  int idx = blockIdx.x * blockDim.x + threadIdx.x;
  if (idx >= cN * cV) return;
  int vv = idx % cV, n = idx / cV;
  float sc[cV];
#pragma unroll 5
  for (int w = 0; w < cV; ++w) {
    float s = 0.f;
    for (int i = 0; i < cINTER; ++i)
      s += th[((size_t)n * cINTER + i) * cV + vv] * ph[((size_t)n * cINTER + i) * cV + w];
    sc[w] = s * 0.125f;     // / sqrt(INTER=64)
  }
  float mx = -1e30f;
#pragma unroll
  for (int w = 0; w < cV; ++w) mx = fmaxf(mx, sc[w]);
  float su = 0.f;
#pragma unroll
  for (int w = 0; w < cV; ++w) { sc[w] = __expf(sc[w] - mx); su += sc[w]; }
  float inv = 1.f / su;
#pragma unroll
  for (int w = 0; w < cV; ++w) E[((size_t)n * cV + vv) * cV + w] = sc[w] * inv;
}

__global__ void aeff_kernel(const float* __restrict__ A, const float* __restrict__ Mm,
                            const float* __restrict__ gam, const float* __restrict__ E,
                            float* __restrict__ Aeff) {
  int idx = blockIdx.x * blockDim.x + threadIdx.x;
  if (idx >= 3 * cN * cV * cV) return;
  int e = idx % (cV * cV), rem = idx / (cV * cV);
  int n = rem % cN, kk = rem / cN;
  Aeff[idx] = A[kk * cV * cV + e] * Mm[kk * cV * cV + e] + gam[0] * E[(size_t)n * cV * cV + e];
}

// agg column-major [NTV][C]
__global__ void agg_kernel(const float* __restrict__ x, const float* __restrict__ Aeff,
                           int kk, half_t* __restrict__ agg) {
  int idx = blockIdx.x * blockDim.x + threadIdx.x;
  if (idx >= cC * cNTV) return;
  int c = idx % cC, col = idx / cC;
  int w = col % cV, nt = col / cV;
  int n = nt / cT, t = nt % cT;
  const float* xp = x + (((size_t)n * cC + c) * cT + t) * cV;
  const float* ae = Aeff + ((size_t)kk * cN + n) * cV * cV + w;
  float s = 0.f;
#pragma unroll
  for (int v = 0; v < cV; ++v) s += ae[v * cV] * xp[v];
  agg[(size_t)col * cC + c] = (half_t)s;
}

__global__ void zerof_kernel(float* __restrict__ p, int n) {
  int idx = blockIdx.x * blockDim.x + threadIdx.x;
  if (idx < n) p[idx] = 0.f;
}

__global__ void bnstats_kernel(const float* __restrict__ gcn, float* __restrict__ stats) {
  __shared__ float s1[256], s2[256];
  int c = blockIdx.x, tid = threadIdx.x;
  float a = 0.f, b = 0.f;
  for (int j = tid; j < cNTV; j += 256) {
    float g = gcn[(size_t)c * cNTV + j];
    a += g; b += g * g;
  }
  s1[tid] = a; s2[tid] = b;
  __syncthreads();
  for (int off = 128; off; off >>= 1) {
    if (tid < off) { s1[tid] += s1[tid + off]; s2[tid] += s2[tid + off]; }
    __syncthreads();
  }
  if (tid == 0) { stats[c] = s1[0]; stats[cDIM + c] = s2[0]; }
}

__global__ void fuse_kernel(const float* __restrict__ x, const float* __restrict__ xtr,
                            const float* __restrict__ gcn, const float* __restrict__ stats,
                            const float* __restrict__ bw, const float* __restrict__ bb,
                            float* __restrict__ out) {
  int idx = blockIdx.x * blockDim.x + threadIdx.x;
  if (idx >= cN * cC * cT * cV) return;
  int vv = idx % cV, rem = idx / cV;
  int t = rem % cT; rem /= cT;
  int c = rem % cC, n = rem / cC;
  size_t j = ((size_t)n * cT + t) * cV + vv;
  float mu  = stats[c] * (1.f / cNTV);
  float var = stats[cDIM + c] * (1.f / cNTV) - mu * mu;
  float g = (gcn[(size_t)c * cNTV + j] - mu) * rsqrtf(var + 1e-5f) * bw[c] + bb[c];
  out[idx] = x[idx] + xtr[(size_t)c * cNTV + j] + fmaxf(g, 0.f);
}

__global__ void cvt_kernel(const float* __restrict__ s, half_t* __restrict__ d, int n) {
  int idx = blockIdx.x * blockDim.x + threadIdx.x;
  if (idx < n) d[idx] = (half_t)s[idx];
}

// ---------------- host-side orchestration ----------------
static inline int div_up(long long a, int b) { return (int)((a + b - 1) / b); }

extern "C" void kernel_launch(void* const* d_in, const int* in_sizes, int n_in,
                              void* d_out, int out_size, void* d_ws, size_t ws_size,
                              hipStream_t stream) {
  const float* x     = (const float*)d_in[0];
  const float* ln_w  = (const float*)d_in[1];
  const float* ln_b  = (const float*)d_in[2];
  const float* W_pe  = (const float*)d_in[3];
  const float* W_v   = (const float*)d_in[4];
  const float* W_k   = (const float*)d_in[5];
  const float* W_q   = (const float*)d_in[6];
  const float* rpe   = (const float*)d_in[7];
  const float* w1    = (const float*)d_in[8];
  const float* outer = (const float*)d_in[9];
  const float* alpha = (const float*)d_in[10];
  const float* Wproj = (const float*)d_in[11];
  const float* bproj = (const float*)d_in[12];
  const float* lq1   = (const float*)d_in[13];
  const float* lk1   = (const float*)d_in[14];
  const float* lq2   = (const float*)d_in[15];
  const float* lk2   = (const float*)d_in[16];
  const float* Mm    = (const float*)d_in[17];
  const float* Wg    = (const float*)d_in[18];
  const float* bg    = (const float*)d_in[19];
  const float* Wth   = (const float*)d_in[20];
  const float* bth   = (const float*)d_in[21];
  const float* Wph   = (const float*)d_in[22];
  const float* bph   = (const float*)d_in[23];
  const float* gamma = (const float*)d_in[24];
  const float* bn_w  = (const float*)d_in[25];
  const float* bn_b  = (const float*)d_in[26];
  const float* Aadj  = (const float*)d_in[27];
  const int*   hops  = (const int*)d_in[28];
  const int*   jl    = (const int*)d_in[29];
  float* out = (float*)d_out;

  char* ws = (char*)d_ws;
  size_t off = 0;
  auto alloc = [&](size_t bytes) -> void* {
    off = (off + 255) & ~(size_t)255;
    void* p = ws + off;
    off += bytes;
    return p;
  };

  half_t* xn16  = (half_t*)alloc((size_t)cNTV * cC * 2);          // column-major [NTV][C]
  half_t* q16   = (half_t*)alloc((size_t)cNT * 2 * cH * cTS * 2); // padded 32x32 tiles
  half_t* k16   = (half_t*)alloc((size_t)cNT * 2 * cH * cTS * 2);
  half_t* v16b  = (half_t*)alloc((size_t)cNT * cH * cTS * 2);
  half_t* ek16  = (half_t*)alloc((size_t)cNT * cH * cTS * 2);
  half_t* z16   = (half_t*)alloc((size_t)cNTG * cC * 2);          // column-major [NTG][C]
  float*  eo    = (float*)alloc((size_t)cDIM * cNTG * 4);
  float*  dterm = (float*)alloc((size_t)cNT * cH * cV * 4);
  float*  xtr   = (float*)alloc((size_t)cDIM * cNTV * 4);
  float*  gcn   = (float*)alloc((size_t)cDIM * cNTV * 4);
  half_t* agg16 = (half_t*)alloc((size_t)cNTV * cC * 2);          // column-major [NTV][C]
  float*  xmean = (float*)alloc((size_t)cN * cC * cV * 4);
  float*  thb   = (float*)alloc((size_t)cN * cINTER * cV * 4);
  float*  phb   = (float*)alloc((size_t)cN * cINTER * cV * 4);
  float*  Ebuf  = (float*)alloc((size_t)cN * cV * cV * 4);
  float*  Aeff  = (float*)alloc((size_t)3 * cN * cV * cV * 4);
  float*  stats = (float*)alloc((size_t)2 * cDIM * 4);
  float*  lam   = (float*)alloc(4 * 4);
  half_t* wq16  = (half_t*)alloc((size_t)512 * cK * 2);
  half_t* wk16  = (half_t*)alloc((size_t)512 * cK * 2);
  half_t* wv16  = (half_t*)alloc((size_t)256 * cK * 2);
  half_t* wpe16 = (half_t*)alloc((size_t)256 * cK * 2);
  half_t* wg16  = (half_t*)alloc((size_t)3 * 256 * cK * 2);
  half_t* wp16  = (half_t*)alloc((size_t)cH * cHD * cHD * 2);
  (void)ws_size; (void)n_in; (void)in_sizes; (void)out_size;

  // weight conversions to f16
  cvt_kernel<<<div_up(512 * cK, 256), 256, 0, stream>>>(W_q, wq16, 512 * cK);
  cvt_kernel<<<div_up(512 * cK, 256), 256, 0, stream>>>(W_k, wk16, 512 * cK);
  cvt_kernel<<<div_up(256 * cK, 256), 256, 0, stream>>>(W_v, wv16, 256 * cK);
  cvt_kernel<<<div_up(256 * cK, 256), 256, 0, stream>>>(W_pe, wpe16, 256 * cK);
  cvt_kernel<<<div_up(3 * 256 * cK, 256), 256, 0, stream>>>(Wg, wg16, 3 * 256 * cK);
  cvt_kernel<<<div_up(cH * cHD * cHD, 256), 256, 0, stream>>>(Wproj, wp16, cH * cHD * cHD);

  // LayerNorm -> xn16 [NTV][C]
  ln_kernel<<<cNTV / 8, 256, 0, stream>>>(x, ln_w, ln_b, xn16);

  // zero the 7 pad rows of every attention tile
  zeropad_kernel<<<div_up((long long)cNT * 2 * cH * 224, 256), 256, 0, stream>>>(q16, cNT * 2 * cH);
  zeropad_kernel<<<div_up((long long)cNT * 2 * cH * 224, 256), 256, 0, stream>>>(k16, cNT * 2 * cH);
  zeropad_kernel<<<div_up((long long)cNT * cH * 224, 256), 256, 0, stream>>>(v16b, cNT * cH);
  zeropad_kernel<<<div_up((long long)cNT * cH * 224, 256), 256, 0, stream>>>(ek16, cNT * cH);

  // Q, K, V projections (WMMA GEMMs, TDM-staged A panel, quad accumulators)
  gemm_wmma_kernel<<<(512 / 128) * (cNTV / 64), 256, 0, stream>>>(
      wq16, xn16, 512, cNTV, 1, nullptr, q16, nullptr);
  gemm_wmma_kernel<<<(512 / 128) * (cNTV / 64), 256, 0, stream>>>(
      wk16, xn16, 512, cNTV, 1, nullptr, k16, nullptr);
  gemm_wmma_kernel<<<(256 / 128) * (cNTV / 64), 256, 0, stream>>>(
      wv16, xn16, 256, cNTV, 2, nullptr, v16b, nullptr);

  // hyperedge token: pool -> pe GEMM -> gather (+d term)
  poolz_kernel<<<div_up((long long)cC * cNTG, 256), 256, 0, stream>>>(x, jl, z16);
  gemm_wmma_kernel<<<(256 / 128) * (cNTG / 64), 256, 0, stream>>>(
      wpe16, z16, 256, cNTG, 0, eo, nullptr, nullptr);
  egather_kernel<<<div_up((long long)cNT * cH * cV, 256), 256, 0, stream>>>(eo, jl, w1, ek16, dterm);

  // lambda, then differential attention + grouped projection
  lam_kernel<<<1, 1, 0, stream>>>(lq1, lk1, lq2, lk2, lam);
  attn_kernel<<<cNT, 256, 0, stream>>>(q16, k16, v16b, ek16, dterm, rpe, hops, outer,
                                       alpha, lam, wp16, bproj, xtr);

  // GCN branch
  xmean_kernel<<<div_up(cN * cC * cV, 256), 256, 0, stream>>>(x, xmean);
  thph_kernel<<<div_up(cN * cINTER * cV, 256), 256, 0, stream>>>(xmean, Wth, bth, Wph, bph, thb, phb);
  scoresE_kernel<<<div_up(cN * cV, 256), 256, 0, stream>>>(thb, phb, Ebuf);
  aeff_kernel<<<div_up(3 * cN * cV * cV, 256), 256, 0, stream>>>(Aadj, Mm, gamma, Ebuf, Aeff);
  zerof_kernel<<<div_up((long long)cDIM * cNTV, 256), 256, 0, stream>>>(gcn, cDIM * cNTV);
  for (int kk = 0; kk < 3; ++kk) {
    agg_kernel<<<div_up((long long)cC * cNTV, 256), 256, 0, stream>>>(x, Aeff, kk, agg16);
    gemm_wmma_kernel<<<(256 / 128) * (cNTV / 64), 256, 0, stream>>>(
        wg16 + (size_t)kk * 256 * cK, agg16, 256, cNTV, 3, gcn, nullptr, bg + kk * cDIM);
  }
  bnstats_kernel<<<cDIM, 256, 0, stream>>>(gcn, stats);

  // fuse: x + x_tr + relu(BN(gcn))
  fuse_kernel<<<div_up((long long)cN * cC * cT * cV, 256), 256, 0, stream>>>(
      x, xtr, gcn, stats, bn_w, bn_b, out);
}
